// adv_transformer_62225486184870
// MI455X (gfx1250) — matrix operand order repaired and checked by reading, not compiled, not run on hardware
//
#include <hip/hip_runtime.h>
#include <hip/hip_bf16.h>
#include <cstdint>

typedef __bf16 bhalf;
typedef __bf16 v16b __attribute__((ext_vector_type(16)));
typedef __bf16 v8b  __attribute__((ext_vector_type(8)));
typedef __bf16 v4b  __attribute__((ext_vector_type(4)));
typedef float  v8f  __attribute__((ext_vector_type(8)));

#define DEPTH     12
#define HEADS     12
#define DIM       768
#define NTOK      196
#define BATCH     32
#define MROWS     (BATCH * NTOK)    /* 6272 */
#define HEAD_DIM  64
#define ATT_SCALE 0.125f            /* 64^-0.5 */
#define SPLICE_M  (BATCH * 3 * 224) /* 21504 */

union frag_u { v16b v; v8b h[2]; };

__device__ inline v8f wmma_bf16(v16b a, v16b b, v8f c) {
  return __builtin_amdgcn_wmma_f32_16x16x32_bf16(false, a, false, b, (short)0, c,
                                                 false, false);
}

// Async copy 16B global -> LDS (gfx1250, tracked with ASYNCcnt)
__device__ inline void async_copy16(uint32_t lds_off, const void* gptr) {
  asm volatile("global_load_async_to_lds_b128 %0, %1, off"
               :: "v"(lds_off), "v"(gptr)
               : "memory");
}
__device__ inline void wait_async0() {
  asm volatile("s_wait_asynccnt 0" ::: "memory");
}

// A fragment (16x32, M x K) from LDS row-major [16][STRIDE]
// lane<16: K 0-7 and 16-23 ; lane>=16: K 8-15 and 24-31 (per ISA layout)
template <int STRIDE>
__device__ inline v16b lds_a_frag(const bhalf* s, int lane) {
  const int m  = lane & 15;
  const int k0 = (lane < 16) ? 0 : 8;
  const bhalf* p = s + m * STRIDE + k0;
  frag_u u;
  u.h[0] = *(const v8b*)p;
  u.h[1] = *(const v8b*)(p + 16);
  return u.v;
}

// B fragment (32x16, K x N) from LDS stored as [n][k] : [16][STRIDE]
// lane<16: col n, K 0-7 and 16-23 ; lane>=16: col n, K 8-15 and 24-31
template <int STRIDE>
__device__ inline v16b lds_b_frag(const bhalf* s, int lane) {
  const int n  = lane & 15;
  const int k0 = (lane < 16) ? 0 : 8;
  const bhalf* p = s + n * STRIDE + k0;
  frag_u u;
  u.h[0] = *(const v8b*)p;
  u.h[1] = *(const v8b*)(p + 16);
  return u.v;
}

// A fragment directly from a global row pointer (row already lane-selected)
__device__ inline v16b g_a_frag(const bhalf* rowp, int lane, int kbase) {
  const int k0 = kbase + ((lane < 16) ? 0 : 8);
  frag_u u;
  u.h[0] = *(const v8b*)(rowp + k0);
  u.h[1] = *(const v8b*)(rowp + k0 + 16);
  return u.v;
}

// B fragment from global where "rows" are N (keys) and K is contiguous per row
__device__ inline v16b g_b_frag(const bhalf* basep, int rowstride, int ntile,
                                int kbase, int lane, int maxrow) {
  int n = ntile + (lane & 15);
  if (n > maxrow) n = maxrow;
  const bhalf* p = basep + (size_t)n * rowstride + kbase + ((lane < 16) ? 0 : 8);
  frag_u u;
  u.h[0] = *(const v8b*)p;
  u.h[1] = *(const v8b*)(p + 16);
  return u.v;
}

// ---------------------------------------------------------------------------
// fp32 -> bf16 weight conversion (one-shot per launch; ~15us of HBM time)
// ---------------------------------------------------------------------------
__global__ __launch_bounds__(256) void cvt_kernel(const float* __restrict__ src,
                                                  bhalf* __restrict__ dst,
                                                  int n4) {
  const int i = blockIdx.x * 256 + threadIdx.x;
  if (i < n4) {
    const float4 f = ((const float4*)src)[i];
    v4b o;
    o[0] = (bhalf)f.x; o[1] = (bhalf)f.y; o[2] = (bhalf)f.z; o[3] = (bhalf)f.w;
    ((v4b*)dst)[i] = o;
  }
}

// ---------------------------------------------------------------------------
// WMMA GEMM: out[M,N] = A[M,K](bf16) @ Wb[N,K]^T(bf16) + bias, with
// double-buffered async-to-LDS staging of both tiles (ASYNCcnt pipeline).
// Epilogue: optional pos-embed broadcast, residual add, exact GELU,
// fp32 and/or bf16 outputs.  M % 128 == 0, K % 32 == 0; N edge handled.
// ---------------------------------------------------------------------------
__global__ __launch_bounds__(256) void gemm_kernel(
    const bhalf* __restrict__ A, const bhalf* __restrict__ Wb,
    const float* __restrict__ bias, const float* __restrict__ pos,
    const float* __restrict__ res, float* __restrict__ outf,
    bhalf* __restrict__ outb, int M, int N, int K, int act) {
  __shared__ __align__(16) bhalf As[2][128][40];
  __shared__ __align__(16) bhalf Bs[2][128][40];
  const int tid  = threadIdx.x;
  const int lane = tid & 31, wave = tid >> 5;
  const int wm = wave >> 2, wn = wave & 3;        // 2 x 4 wave grid
  const int mblk = blockIdx.y * 128, nblk = blockIdx.x * 128;
  v8f acc[4][2] = {};

  // 128x32 tile = 512 chunks of 8 bf16 (16B); each thread stages 2 chunks of
  // A and 2 of B per k-step, fully async into LDS.
  const int ch0 = tid * 2;
  auto stage = [&](int buf, int kt) {
#pragma unroll
    for (int q = 0; q < 2; ++q) {
      const int ch = ch0 + q;
      const int rr = ch >> 2, seg = (ch & 3) * 8;
      async_copy16((uint32_t)(uintptr_t)&As[buf][rr][seg],
                   A + (size_t)(mblk + rr) * K + kt + seg);
      int gn = nblk + rr;
      if (gn > N - 1) gn = N - 1;
      async_copy16((uint32_t)(uintptr_t)&Bs[buf][rr][seg],
                   Wb + (size_t)gn * K + kt + seg);
    }
  };

  stage(0, 0);
  wait_async0();
  __syncthreads();

  int buf = 0;
  for (int kt = 0; kt < K; kt += 32) {
    if (kt + 32 < K) stage(buf ^ 1, kt + 32);   // prefetch next tiles (async)

    v16b af[4], bfr[2];
#pragma unroll
    for (int i = 0; i < 4; ++i)
      af[i] = lds_a_frag<40>(&As[buf][wm * 64 + i * 16][0], lane);
#pragma unroll
    for (int j = 0; j < 2; ++j)
      bfr[j] = lds_b_frag<40>(&Bs[buf][wn * 32 + j * 16][0], lane);
#pragma unroll
    for (int i = 0; i < 4; ++i)
#pragma unroll
      for (int j = 0; j < 2; ++j)
        acc[i][j] = wmma_bf16(af[i], bfr[j], acc[i][j]);

    wait_async0();        // next-tile copies landed
    __syncthreads();      // everyone done reading current tile
    buf ^= 1;
  }

  // epilogue: C layout -> row = r + (lane<16?0:8), col = lane&15
  const int mw = mblk + wm * 64, nw = nblk + wn * 32;
#pragma unroll
  for (int i = 0; i < 4; ++i)
#pragma unroll
    for (int j = 0; j < 2; ++j)
#pragma unroll
      for (int r = 0; r < 8; ++r) {
        const int row = mw + i * 16 + r + (lane < 16 ? 0 : 8);
        const int col = nw + j * 16 + (lane & 15);
        if (col < N) {
          float v = acc[i][j][r] + bias[col];
          if (pos)  v += pos[(size_t)(row % NTOK) * N + col];
          if (res)  v += res[(size_t)row * N + col];
          if (act)  v = 0.5f * v * (1.0f + erff(v * 0.70710678118654752f));
          if (outf) outf[(size_t)row * N + col] = v;
          if (outb) outb[(size_t)row * N + col] = (bhalf)v;
        }
      }
}

// ---------------------------------------------------------------------------
// LayerNorm over DIM=768 (one row per block, 256 threads x 3 elems) -> bf16
// ---------------------------------------------------------------------------
__global__ __launch_bounds__(256) void ln_kernel(const float* __restrict__ h,
                                                 const float* __restrict__ w,
                                                 const float* __restrict__ b,
                                                 bhalf* __restrict__ out) {
  const int row = blockIdx.x;
  const int tid = threadIdx.x;
  const float* x = h + (size_t)row * DIM;
  float vals[3], s = 0.f, s2 = 0.f;
#pragma unroll
  for (int i = 0; i < 3; ++i) {
    float v = x[i * 256 + tid];
    vals[i] = v;
    s += v;
    s2 += v * v;
  }
  __shared__ float rs[8], rs2[8], stat[2];
  float a = s, a2 = s2;
#pragma unroll
  for (int o = 16; o; o >>= 1) {
    a  += __shfl_xor(a, o, 32);
    a2 += __shfl_xor(a2, o, 32);
  }
  if ((tid & 31) == 0) { rs[tid >> 5] = a; rs2[tid >> 5] = a2; }
  __syncthreads();
  if (tid == 0) {
    float ta = 0.f, tb = 0.f;
#pragma unroll
    for (int i = 0; i < 8; ++i) { ta += rs[i]; tb += rs2[i]; }
    float mu  = ta * (1.0f / DIM);
    float var = tb * (1.0f / DIM) - mu * mu;
    stat[0] = mu;
    stat[1] = rsqrtf(var + 1e-5f);
  }
  __syncthreads();
  const float mu = stat[0], inv = stat[1];
#pragma unroll
  for (int i = 0; i < 3; ++i) {
    int c = i * 256 + tid;
    out[(size_t)row * DIM + c] = (bhalf)((vals[i] - mu) * inv * w[c] + b[c]);
  }
}

// ---------------------------------------------------------------------------
// Flash-style attention with WMMA.  One block per (batch, head).
// qkv: bf16 [MROWS, 2304] laid out [.., 3, HEADS, 64]; o: bf16 [MROWS, 768]
// ---------------------------------------------------------------------------
__global__ __launch_bounds__(256) void attn_kernel(const bhalf* __restrict__ qkv,
                                                   const float* __restrict__ relb,
                                                   bhalf* __restrict__ o) {
  const int bb = blockIdx.x / HEADS;
  const int hd = blockIdx.x % HEADS;
  const int tid = threadIdx.x, lane = tid & 31, wave = tid >> 5;
  const bhalf* base = qkv + (size_t)bb * NTOK * (3 * DIM);

  __shared__ __align__(16) bhalf Vt[64][240];      // V^T : [dim][key]
  __shared__ __align__(16) bhalf Pb[8][16][32];    // per-wave P re-layout

  // stage V transposed (zero-pad keys >= 196)
  for (int idx = tid; idx < 224 * 64; idx += 256) {
    const int key = idx >> 6, d = idx & 63;
    bhalf v = (bhalf)0.0f;
    if (key < NTOK) v = base[(size_t)key * (3 * DIM) + 2 * DIM + hd * 64 + d];
    Vt[d][key] = v;
  }
  __syncthreads();

  const bhalf* kbase_p = base + DIM + hd * 64;     // K matrix base

  for (int t = wave; t < 14; t += 8) {             // 14 query tiles of 16
    const int q0 = t * 16;
    int qr = q0 + (lane & 15);
    if (qr > NTOK - 1) qr = NTOK - 1;
    const bhalf* qrow = base + (size_t)qr * (3 * DIM) + hd * 64;
    const v16b a0 = g_a_frag(qrow, lane, 0);
    const v16b a1 = g_a_frag(qrow, lane, 32);

    v8f oacc[4] = {};
    float mrow[8], srow[8];
#pragma unroll
    for (int r = 0; r < 8; ++r) { mrow[r] = -1e30f; srow[r] = 0.f; }

    for (int kk = 0; kk < 7; ++kk) {               // 7 x 32 keys = 224 (padded)
      const int key0 = kk * 32;
      v8f s0 = {}, s1 = {};
      s0 = wmma_bf16(a0, g_b_frag(kbase_p, 3 * DIM, key0,       0, lane, NTOK - 1), s0);
      s0 = wmma_bf16(a1, g_b_frag(kbase_p, 3 * DIM, key0,      32, lane, NTOK - 1), s0);
      s1 = wmma_bf16(a0, g_b_frag(kbase_p, 3 * DIM, key0 + 16,  0, lane, NTOK - 1), s1);
      s1 = wmma_bf16(a1, g_b_frag(kbase_p, 3 * DIM, key0 + 16, 32, lane, NTOK - 1), s1);

      const int qbase = q0 + (lane < 16 ? 0 : 8);
#pragma unroll
      for (int r = 0; r < 8; ++r) {
        int qq = qbase + r;
        if (qq > NTOK - 1) qq = NTOK - 1;
        const int qy = qq / 14, qx = qq - qy * 14;
        const int c0 = key0 + (lane & 15);
        const int c1 = c0 + 16;
        const int cc0 = (c0 > NTOK - 1) ? NTOK - 1 : c0;
        const int cc1 = (c1 > NTOK - 1) ? NTOK - 1 : c1;
        const int ky0 = cc0 / 14, kx0 = cc0 - ky0 * 14;
        const int ky1 = cc1 / 14, kx1 = cc1 - ky1 * 14;
        float v0 = s0[r] * ATT_SCALE +
                   relb[((qy - ky0 + 13) * 27 + (qx - kx0 + 13)) * HEADS + hd];
        float v1 = s1[r] * ATT_SCALE +
                   relb[((qy - ky1 + 13) * 27 + (qx - kx1 + 13)) * HEADS + hd];
        if (c0 > NTOK - 1) v0 = -1e30f;
        if (c1 > NTOK - 1) v1 = -1e30f;
        float mx = fmaxf(v0, v1);
#pragma unroll
        for (int o2 = 8; o2; o2 >>= 1) mx = fmaxf(mx, __shfl_xor(mx, o2, 32));
        const float mnew = fmaxf(mrow[r], mx);
        const float sf = __expf(mrow[r] - mnew);
        mrow[r] = mnew;
        const float p0 = __expf(v0 - mnew);
        const float p1 = __expf(v1 - mnew);
        float rsum = p0 + p1;
#pragma unroll
        for (int o2 = 8; o2; o2 >>= 1) rsum += __shfl_xor(rsum, o2, 32);
        srow[r] = srow[r] * sf + rsum;
#pragma unroll
        for (int j = 0; j < 4; ++j) oacc[j][r] *= sf;
        // spill P (C layout) to LDS for A-layout reload
        const int prow = r + (lane < 16 ? 0 : 8);
        Pb[wave][prow][(lane & 15)]      = (bhalf)p0;
        Pb[wave][prow][16 + (lane & 15)] = (bhalf)p1;
      }
      asm volatile("s_wait_dscnt 0" ::: "memory");   // wave-local LDS RAW
      const v16b pa = lds_a_frag<32>(&Pb[wave][0][0], lane);
#pragma unroll
      for (int j = 0; j < 4; ++j) {                  // O += P @ V
        const bhalf* vp = &Vt[j * 16 + (lane & 15)][key0 + ((lane < 16) ? 0 : 8)];
        frag_u u;
        u.h[0] = *(const v8b*)vp;
        u.h[1] = *(const v8b*)(vp + 16);
        oacc[j] = wmma_bf16(pa, u.v, oacc[j]);
      }
    }

#pragma unroll
    for (int j = 0; j < 4; ++j)
#pragma unroll
      for (int r = 0; r < 8; ++r) {
        const int row = q0 + r + (lane < 16 ? 0 : 8);
        if (row < NTOK) {
          const float v = oacc[j][r] / srow[r];
          o[((size_t)(bb * NTOK + row)) * DIM + hd * 64 + j * 16 + (lane & 15)] =
              (bhalf)v;
        }
      }
  }
}

// ---------------------------------------------------------------------------
// Patch gather: x[B,3,224,224] -> bf16 A [6272, 768]
// ---------------------------------------------------------------------------
__global__ __launch_bounds__(256) void patch_gather_kernel(
    const float* __restrict__ x, bhalf* __restrict__ A) {
  const int idx = blockIdx.x * 256 + threadIdx.x;      // exact grid
  const int row = idx / DIM, col = idx - row * DIM;
  const int b = row / NTOK, n = row - b * NTOK;
  const int gy = n / 14, gx = n - gy * 14;
  const int c = col >> 8, p = col & 255, py = p >> 4, px = p & 15;
  A[idx] = (bhalf)x[(((size_t)b * 3 + c) * 224 + gy * 16 + py) * 224 +
                    gx * 16 + px];
}

// ---------------------------------------------------------------------------
// Splice gather: h fp32 [6272,768] -> bf16 Z [21504, 224]
// ---------------------------------------------------------------------------
__global__ __launch_bounds__(256) void splice_gather_kernel(
    const float* __restrict__ h, bhalf* __restrict__ Z) {
  const int idx = blockIdx.x * 256 + threadIdx.x;      // exact grid
  const int row = idx / 224, col = idx - row * 224;
  const int b = row / 672, rc = row - b * 672;
  const int c = rc / 224, rr = rc - c * 224;
  const int gy = rr >> 4, py = rr & 15;
  const int gx = col >> 4, px = col & 15;
  Z[idx] = (bhalf)h[((size_t)b * NTOK + gy * 14 + gx) * DIM + c * 256 +
                    py * 16 + px];
}

// ---------------------------------------------------------------------------
extern "C" void kernel_launch(void* const* d_in, const int* in_sizes, int n_in,
                              void* d_out, int out_size, void* d_ws,
                              size_t ws_size, hipStream_t stream) {
  const float* x        = (const float*)d_in[0];
  const float* patch_w  = (const float*)d_in[1];
  const float* patch_b  = (const float*)d_in[2];
  const float* pos_emb  = (const float*)d_in[3];
  const float* norm_w   = (const float*)d_in[4];
  const float* norm_b   = (const float*)d_in[5];
  const float* qkv_w    = (const float*)d_in[6];
  const float* qkv_b    = (const float*)d_in[7];
  const float* rel_bias = (const float*)d_in[8];
  const float* proj_w   = (const float*)d_in[9];
  const float* proj_b   = (const float*)d_in[10];
  const float* fc1_w    = (const float*)d_in[11];
  const float* fc1_b    = (const float*)d_in[12];
  const float* fc2_w    = (const float*)d_in[13];
  const float* fc2_b    = (const float*)d_in[14];
  const float* splice_w = (const float*)d_in[15];
  const float* splice_b = (const float*)d_in[16];
  (void)in_sizes; (void)n_in; (void)out_size; (void)ws_size;

  char* wsb = (char*)d_ws;
  size_t off = 0;
  auto carve = [&](size_t bytes) -> void* {
    void* p = wsb + off;
    off += (bytes + 255) & ~(size_t)255;
    return p;
  };
  float* h      = (float*)carve((size_t)MROWS * DIM * 4);
  bhalf* hn     = (bhalf*)carve((size_t)MROWS * DIM * 2);
  bhalf* qkv    = (bhalf*)carve((size_t)MROWS * 3 * DIM * 2);
  bhalf* attno  = (bhalf*)carve((size_t)MROWS * DIM * 2);
  bhalf* mid    = (bhalf*)carve((size_t)MROWS * 4 * DIM * 2);
  bhalf* apatch = (bhalf*)carve((size_t)MROWS * DIM * 2);
  bhalf* zspl   = (bhalf*)carve((size_t)SPLICE_M * 224 * 2);
  // bf16 weights
  const size_t n_qkvw = (size_t)DEPTH * 3 * DIM * DIM;
  const size_t n_prjw = (size_t)DEPTH * DIM * DIM;
  const size_t n_fc1w = (size_t)DEPTH * 4 * DIM * DIM;
  const size_t n_fc2w = (size_t)DEPTH * DIM * 4 * DIM;
  const size_t n_patw = (size_t)DIM * DIM;
  const size_t n_splw = (size_t)224 * 224;
  bhalf* qkv_wb    = (bhalf*)carve(n_qkvw * 2);
  bhalf* proj_wb   = (bhalf*)carve(n_prjw * 2);
  bhalf* fc1_wb    = (bhalf*)carve(n_fc1w * 2);
  bhalf* fc2_wb    = (bhalf*)carve(n_fc2w * 2);
  bhalf* patch_wb  = (bhalf*)carve(n_patw * 2);
  bhalf* splice_wb = (bhalf*)carve(n_splw * 2);

  const dim3 blk(256);
  auto cvt = [&](const float* s, bhalf* d, size_t n) {
    const int n4 = (int)(n / 4);
    cvt_kernel<<<(n4 + 255) / 256, blk, 0, stream>>>(s, d, n4);
  };
  cvt(qkv_w, qkv_wb, n_qkvw);
  cvt(proj_w, proj_wb, n_prjw);
  cvt(fc1_w, fc1_wb, n_fc1w);
  cvt(fc2_w, fc2_wb, n_fc2w);
  cvt(patch_w, patch_wb, n_patw);
  cvt(splice_w, splice_wb, n_splw);

  // Patch embedding: gather + GEMM(+bias+pos_embed) -> h fp32
  patch_gather_kernel<<<(MROWS * DIM) / 256, blk, 0, stream>>>(x, apatch);
  gemm_kernel<<<dim3(DIM / 128, MROWS / 128), blk, 0, stream>>>(
      apatch, patch_wb, patch_b, pos_emb, nullptr, h, nullptr, MROWS, DIM, DIM, 0);

  for (int i = 0; i < DEPTH; ++i) {
    ln_kernel<<<MROWS, blk, 0, stream>>>(h, norm_w + i * DIM, norm_b + i * DIM, hn);
    gemm_kernel<<<dim3(3 * DIM / 128, MROWS / 128), blk, 0, stream>>>(
        hn, qkv_wb + (size_t)i * 3 * DIM * DIM, qkv_b + (size_t)i * 3 * DIM,
        nullptr, nullptr, nullptr, qkv, MROWS, 3 * DIM, DIM, 0);
    attn_kernel<<<BATCH * HEADS, blk, 0, stream>>>(
        qkv, rel_bias + (size_t)i * 729 * HEADS, attno);
    gemm_kernel<<<dim3(DIM / 128, MROWS / 128), blk, 0, stream>>>(
        attno, proj_wb + (size_t)i * DIM * DIM, proj_b + (size_t)i * DIM,
        nullptr, h, h, nullptr, MROWS, DIM, DIM, 0);

    ln_kernel<<<MROWS, blk, 0, stream>>>(h, norm_w + i * DIM, norm_b + i * DIM, hn);
    gemm_kernel<<<dim3(4 * DIM / 128, MROWS / 128), blk, 0, stream>>>(
        hn, fc1_wb + (size_t)i * 4 * DIM * DIM, fc1_b + (size_t)i * 4 * DIM,
        nullptr, nullptr, nullptr, mid, MROWS, 4 * DIM, DIM, 1);
    gemm_kernel<<<dim3(DIM / 128, MROWS / 128), blk, 0, stream>>>(
        mid, fc2_wb + (size_t)i * DIM * 4 * DIM, fc2_b + (size_t)i * DIM,
        nullptr, h, h, nullptr, MROWS, DIM, 4 * DIM, 0);
  }

  // Patch splicing: gather + GEMM over width -> d_out fp32
  splice_gather_kernel<<<(SPLICE_M * 224) / 256, blk, 0, stream>>>(h, zspl);
  gemm_kernel<<<dim3(2, SPLICE_M / 128), blk, 0, stream>>>(
      zspl, splice_wb, splice_b, nullptr, nullptr, (float*)d_out, nullptr,
      SPLICE_M, 224, 224, 0);
}
